// StateEliminationNNet_17695265259706
// MI455X (gfx1250) — matrix operand order repaired and checked by reading, not compiled, not run on hardware
//
#include <hip/hip_runtime.h>
#include <hip/hip_bf16.h>
#include <math.h>

// ---------------------------------------------------------------------------
// Types for CDNA5 WMMA
// ---------------------------------------------------------------------------
typedef _Float16 half_t;
typedef __attribute__((ext_vector_type(16))) _Float16 v16h;
typedef __attribute__((ext_vector_type(8)))  _Float16 v8h;
typedef __attribute__((ext_vector_type(8)))  float    v8f;

// ---------------------------------------------------------------------------
// Problem constants (from reference)
// ---------------------------------------------------------------------------
#define G_      64
#define P_      50
#define N_      3200          // G*P
#define E_      25600         // N*DEG
#define NE_     28800         // E + N (self loops appended)
#define VOCAB_  128
#define A_      50
#define SEMB_   42
#define REMB_   64
#define LSTMD_  64
#define CH_     256
#define HDS_    4
#define HC_     64
#define NEG_SLOPE 0.2f

// ---------------------------------------------------------------------------
// Helpers
// ---------------------------------------------------------------------------
__device__ __forceinline__ float sigmoidf_(float x) { return 1.0f / (1.0f + __expf(-x)); }

// order-preserving float <-> uint mapping for atomicMax on floats
__device__ __forceinline__ unsigned ordf(float f) {
    unsigned u = __float_as_uint(f);
    return (u & 0x80000000u) ? ~u : (u | 0x80000000u);
}
__device__ __forceinline__ float deordf(unsigned u) {
    return __uint_as_float((u & 0x80000000u) ? (u & 0x7fffffffu) : ~u);
}

__device__ __forceinline__ v16h pack16(v8h lo, v8h hi) {
    v16h r;
#pragma unroll
    for (int i = 0; i < 8; ++i) { r[i] = lo[i]; r[i + 8] = hi[i]; }
    return r;
}

// A-matrix fragment (16x32 f16): lane&15 = M row; lanes 0-15 hold K = 0..15,
// lanes 16-31 hold K = 16..31. 'row' points at A[M][k0].
__device__ __forceinline__ v16h load_a_frag(const half_t* row, int g) {
    v8h lo = *(const v8h*)(row + g * 16);
    v8h hi = *(const v8h*)(row + g * 16 + 8);
    return pack16(lo, hi);
}

// B-matrix fragment (32x16 f16): lane&15 = N col; K = g*16 + s (s=0..15 contiguous).
// 'wrow' points at W[N][k0 + g*16] (W stored row-major [Nout][K], B[k][n] = W[n][k]).
__device__ __forceinline__ v16h load_b_frag(const half_t* wrow) {
    v8h lo = *(const v8h*)(wrow);
    v8h hi = *(const v8h*)(wrow + 8);
    return pack16(lo, hi);
}

// ---------------------------------------------------------------------------
// Utility kernels
// ---------------------------------------------------------------------------
__global__ void k_cvt_f16(const float* __restrict__ s, half_t* __restrict__ d, int n) {
    int i = blockIdx.x * blockDim.x + threadIdx.x;
    if (i < n) d[i] = (half_t)s[i];
}
__global__ void k_zero_f32(float* p, int n) {
    int i = blockIdx.x * blockDim.x + threadIdx.x;
    if (i < n) p[i] = 0.0f;
}
__global__ void k_zero_u32(unsigned* p, int n) {
    int i = blockIdx.x * blockDim.x + threadIdx.x;
    if (i < n) p[i] = 0u;
}

// ---------------------------------------------------------------------------
// LSTM encoder: 16-step scan over E sequences.  One wave per 16-edge tile,
// 8 waves / block.  Gates GEMM done with v_wmma_f32_16x16x32_f16.
//
// ALL per-step operands come from LDS (embedding table, weights, h state):
// the step loop contains __syncthreads(), so the compiler cannot hoist the
// weight-fragment loads out of the loop.  Gate accumulators start at ZERO
// (inline-0 C operand) and the bias is folded into the elementwise activation
// stage -- keeping 128 VGPRs of loop-invariant bias-broadcast values out of
// the live set (previous build ran >256 VGPRs because of them).
// LDS: 16 KB embed + 32 KB w_ih + 32 KB w_hh + 32 KB h double-buffer = 112 KB.
// ---------------------------------------------------------------------------
__global__ void __launch_bounds__(256) k_lstm(
    const int*   __restrict__ edge_attr,    // [E][18], cols 0..15 tokens
    const float* __restrict__ regex_embed,  // [128][64]
    const float* __restrict__ b_ih,         // [256]
    const float* __restrict__ b_hh,         // [256]
    const float* __restrict__ w_ih,         // [256][64] f32 (converted to LDS f16)
    const float* __restrict__ w_hh,         // [256][64] f32 (converted to LDS f16)
    float* __restrict__ enc,                // [E][64]
    half_t* __restrict__ enc16)             // [E][64]
{
    __shared__ half_t s_emb[VOCAB_ * REMB_];   // 16 KB
    __shared__ half_t s_wih[CH_ * REMB_];      // 32 KB
    __shared__ half_t s_whh[CH_ * REMB_];      // 32 KB
    __shared__ half_t s_h[8][2][16 * 64];      // 32 KB (double buffered per wave)

    const int tid = threadIdx.x;
    for (int i = tid; i < VOCAB_ * REMB_; i += 256) s_emb[i] = (half_t)regex_embed[i];
    for (int i = tid; i < CH_ * REMB_; i += 256) {
        s_wih[i] = (half_t)w_ih[i];
        s_whh[i] = (half_t)w_hh[i];
    }

    const int w    = tid >> 5;
    const int lane = tid & 31;
    const int ln   = lane & 15;
    const int g    = lane >> 4;
    const int base = blockIdx.x * 128 + w * 16;   // first edge of this wave's tile
    const int erow = base + ln;

    // zero initial h buffer (buffer 0)
    for (int i = lane; i < 1024; i += 32) s_h[w][0][i] = (half_t)0.0f;

    // combined bias for the 16 column-tiles this lane touches
    float bias_l[16];
#pragma unroll
    for (int j = 0; j < 16; ++j) bias_l[j] = b_ih[j * 16 + ln] + b_hh[j * 16 + ln];

    // cell state, D-fragment layout, 4 tiles of 16 hidden cols each
    v8f cst[4];
#pragma unroll
    for (int j = 0; j < 4; ++j)
#pragma unroll
        for (int r = 0; r < 8; ++r) cst[j][r] = 0.0f;

    __syncthreads();

#pragma unroll 1
    for (int t = 0; t < 16; ++t) {
        const int pr = t & 1;       // read buffer
        const int pw = pr ^ 1;      // write buffer
        const int tok = edge_attr[erow * 18 + t];
        const half_t* xrow = &s_emb[tok * 64];
        const half_t* hrow = &s_h[w][pr][ln * 64];
        half_t*       hout = &s_h[w][pw][0];

        v16h ax0 = load_a_frag(xrow,      g);
        v16h ax1 = load_a_frag(xrow + 32, g);
        v16h ah0 = load_a_frag(hrow,      g);
        v16h ah1 = load_a_frag(hrow + 32, g);

#pragma unroll
        for (int j = 0; j < 4; ++j) {               // hidden-column tile (16 cols)
            // zero-init accumulators (backend can use inline-0 SRC2)
            v8f gi, gf, gg, go;
#pragma unroll
            for (int r = 0; r < 8; ++r) { gi[r] = 0.0f; gf[r] = 0.0f; gg[r] = 0.0f; go[r] = 0.0f; }
#pragma unroll
            for (int kf = 0; kf < 2; ++kf) {
                v16h ax = kf ? ax1 : ax0;
                v16h ah = kf ? ah1 : ah0;
                const int kofs = kf * 32 + g * 16;
#define GATE_MMA(accv, tj)                                                                   \
    accv = __builtin_amdgcn_wmma_f32_16x16x32_f16(false, ax, false,                           \
            load_b_frag(s_wih + ((tj) * 16 + ln) * 64 + kofs), (short)0, accv, false, false); \
    accv = __builtin_amdgcn_wmma_f32_16x16x32_f16(false, ah, false,                           \
            load_b_frag(s_whh + ((tj) * 16 + ln) * 64 + kofs), (short)0, accv, false, false);
                GATE_MMA(gi, j)
                GATE_MMA(gf, j + 4)
                GATE_MMA(gg, j + 8)
                GATE_MMA(go, j + 12)
#undef GATE_MMA
            }
            // elementwise LSTM cell in D-fragment layout (bias folded in here):
            // lane: col = j*16 + ln ; vgpr r: row M = g*8 + r
#pragma unroll
            for (int r = 0; r < 8; ++r) {
                float iv = sigmoidf_(gi[r] + bias_l[j]);
                float fv = sigmoidf_(gf[r] + bias_l[j + 4]);
                float gv = tanhf    (gg[r] + bias_l[j + 8]);
                float ov = sigmoidf_(go[r] + bias_l[j + 12]);
                float cv = fv * cst[j][r] + iv * gv;
                cst[j][r] = cv;
                float hv = ov * tanhf(cv);
                hout[(g * 8 + r) * 64 + j * 16 + ln] = (half_t)hv;
            }
        }
        __syncthreads();
    }

    // final h lives in buffer 0 (after 16 steps)
    const half_t* hf = &s_h[w][0][0];
    for (int i = lane; i < 1024; i += 32) {
        int row = i >> 6, col = i & 63;
        float v = (float)hf[i];
        enc  [(base + row) * 64 + col] = v;
        enc16[(base + row) * 64 + col] = (half_t)v;
    }
}

// ---------------------------------------------------------------------------
// Generic WMMA GEMM:  C[M][Nout] = A16[M][K] @ W16[Nout][K]^T + bias[Nout]
// One 16x16 C tile per wave; 8 waves per block.
// ---------------------------------------------------------------------------
__global__ void __launch_bounds__(256) k_gemm(
    const half_t* __restrict__ A, const half_t* __restrict__ W,
    const float* __restrict__ bias, float* __restrict__ C,
    int M, int K, int Nout)
{
    const int lane   = threadIdx.x & 31;
    const int wave   = threadIdx.x >> 5;
    const int ntiles = Nout >> 4;
    const int total  = (M >> 4) * ntiles;
    const int tile   = blockIdx.x * 8 + wave;
    if (tile >= total) return;
    const int mt = tile / ntiles;
    const int nt = tile % ntiles;
    const int ln = lane & 15;
    const int g  = lane >> 4;

    v8f acc;
    float b0 = bias ? bias[nt * 16 + ln] : 0.0f;
#pragma unroll
    for (int r = 0; r < 8; ++r) acc[r] = b0;

    const half_t* arow = A + (size_t)(mt * 16 + ln) * K;
    const half_t* wrow = W + (size_t)(nt * 16 + ln) * K + g * 16;

    for (int kk = 0; kk < K; kk += 32) {
        __builtin_prefetch(arow + kk + 64, 0, 1);
        v16h a = load_a_frag(arow + kk, g);
        v16h b = load_b_frag(wrow + kk);
        acc = __builtin_amdgcn_wmma_f32_16x16x32_f16(false, a, false, b, (short)0,
                                                     acc, false, false);
    }
    const int n  = nt * 16 + ln;
    const int mb = mt * 16 + g * 8;
#pragma unroll
    for (int r = 0; r < 8; ++r) C[(size_t)(mb + r) * Nout + n] = acc[r];
}

// ---------------------------------------------------------------------------
// Edge-parallel pooling accumulators (mean pools + loop_attr numerator)
// ---------------------------------------------------------------------------
__global__ void k_pool(const int* __restrict__ ea, const int* __restrict__ src,
                       const int* __restrict__ tgt, const float* __restrict__ semb,
                       const float* __restrict__ enc,
                       float* oa, float* ia, float* co, float* ci, float* laacc)
{
    int e = blockIdx.x * blockDim.x + threadIdx.x;
    if (e >= E_) return;
    int s  = src[e], t = tgt[e];
    int ss = ea[e * 18 + 16];   // edge_attr[:, -2] -> src state
    int st = ea[e * 18 + 17];   // edge_attr[:, -1] -> tgt state
    atomicAdd(&co[s], 1.0f);
    atomicAdd(&ci[t], 1.0f);
    for (int d = 0; d < SEMB_; ++d) {
        atomicAdd(&oa[s * 106 + d], semb[st * SEMB_ + d]);   // tgt_emb -> out_tr
        atomicAdd(&ia[t * 106 + d], semb[ss * SEMB_ + d]);   // src_emb -> in_tr
    }
    for (int d = 0; d < 64; ++d) {
        float v = enc[e * 64 + d];
        atomicAdd(&oa[s * 106 + SEMB_ + d], v);
        atomicAdd(&ia[t * 106 + SEMB_ + d], v);
        atomicAdd(&laacc[t * 64 + d], v);
    }
}

__global__ void k_build_h(const int* __restrict__ x, const float* __restrict__ semb,
                          const float* __restrict__ oa, const float* __restrict__ ia,
                          const float* __restrict__ co, const float* __restrict__ ci,
                          float* __restrict__ h, half_t* __restrict__ h16)
{
    int i = blockIdx.x * blockDim.x + threadIdx.x;
    if (i >= N_) return;
    float din  = fmaxf(ci[i], 1.0f);
    float dout = fmaxf(co[i], 1.0f);
    float*  hr   = h   + (size_t)i * 256;
    half_t* hr16 = h16 + (size_t)i * 256;
    int st = x[i * 3];
    for (int d = 0; d < SEMB_; ++d) hr[d] = semb[st * SEMB_ + d];
    hr[42] = (float)x[i * 3 + 1];
    hr[43] = (float)x[i * 3 + 2];
    for (int d = 0; d < 106; ++d) {
        hr[44 + d]  = ia[i * 106 + d] / din;
        hr[150 + d] = oa[i * 106 + d] / dout;
    }
    for (int d = 0; d < 256; ++d) hr16[d] = (half_t)hr[d];
}

__global__ void k_build_ea(const half_t* __restrict__ enc16,
                           const float* __restrict__ laacc,
                           const float* __restrict__ ci, half_t* __restrict__ ea16)
{
    int i = blockIdx.x * blockDim.x + threadIdx.x;
    if (i < E_ * 64) {
        ea16[i] = enc16[i];
    } else if (i < NE_ * 64) {
        int j = i - E_ * 64;
        int nidx = j >> 6;
        ea16[i] = (half_t)(laacc[j] / fmaxf(ci[nidx], 1.0f));
    }
}

// ---------------------------------------------------------------------------
// GATv2 attention: score -> segment max -> exp/sum -> aggregate
// ---------------------------------------------------------------------------
__global__ void k_score(const int* __restrict__ src, const int* __restrict__ tgt,
                        const float* __restrict__ xl, const float* __restrict__ xr,
                        const float* __restrict__ ee, const float* __restrict__ att,
                        float* __restrict__ score, unsigned* __restrict__ smax)
{
    int e = blockIdx.x * blockDim.x + threadIdx.x;
    if (e >= NE_) return;
    int s, t;
    if (e < E_) { s = src[e]; t = tgt[e]; } else { s = t = e - E_; }
    const float* xls = xl + (size_t)s * 256;
    const float* xrt = xr + (size_t)t * 256;
    const float* eer = ee + (size_t)e * 256;
    for (int hd = 0; hd < HDS_; ++hd) {
        float sc = 0.0f;
        for (int c = 0; c < HC_; ++c) {
            int cc = hd * HC_ + c;
            float z = xls[cc] + xrt[cc] + eer[cc];
            z = z > 0.0f ? z : NEG_SLOPE * z;
            sc += z * att[hd * HC_ + c];
        }
        score[e * 4 + hd] = sc;
        atomicMax(&smax[t * 4 + hd], ordf(sc));
    }
}

__global__ void k_expsum(const int* __restrict__ tgt, const float* __restrict__ score,
                         const unsigned* __restrict__ smax,
                         float* __restrict__ alpha, float* __restrict__ denom)
{
    int e = blockIdx.x * blockDim.x + threadIdx.x;
    if (e >= NE_) return;
    int t = (e < E_) ? tgt[e] : (e - E_);
    for (int hd = 0; hd < HDS_; ++hd) {
        float wv = __expf(score[e * 4 + hd] - deordf(smax[t * 4 + hd]));
        alpha[e * 4 + hd] = wv;
        atomicAdd(&denom[t * 4 + hd], wv);
    }
}

__global__ void k_agg(const int* __restrict__ src, const int* __restrict__ tgt,
                      const float* __restrict__ xl, const float* __restrict__ alpha,
                      const float* __restrict__ denom, float* __restrict__ agg)
{
    int e = blockIdx.x * blockDim.x + threadIdx.x;
    if (e >= NE_) return;
    int s, t;
    if (e < E_) { s = src[e]; t = tgt[e]; } else { s = t = e - E_; }
    float al[4];
    for (int hd = 0; hd < HDS_; ++hd) al[hd] = alpha[e * 4 + hd] / denom[t * 4 + hd];
    const float* xls = xl + (size_t)s * 256;
    for (int cc = 0; cc < 256; ++cc)
        atomicAdd(&agg[(size_t)t * 256 + cc], al[cc >> 6] * xls[cc]);
}

__global__ void k_layer_final(const float* __restrict__ agg, const float* __restrict__ cb,
                              float* __restrict__ h, half_t* __restrict__ h16)
{
    int i = blockIdx.x * blockDim.x + threadIdx.x;
    if (i >= N_ * 256) return;
    int c = i & 255;
    float v = agg[i] + cb[c];
    v = v > 0.0f ? v : 0.0f;
    v += h[i];
    h[i]  = v;
    h16[i] = (half_t)v;
}

// ---------------------------------------------------------------------------
// Heads
// ---------------------------------------------------------------------------
__global__ void k_gpool(const float* __restrict__ h, float* __restrict__ sg)
{
    int g = blockIdx.x, c = threadIdx.x;   // <<<G_, 256>>>
    float s = 0.0f;
    for (int p = 0; p < P_; ++p) s += h[(size_t)(g * P_ + p) * 256 + c];
    sg[g * 256 + c] = s / (float)P_;
}

__global__ void k_value(const float* __restrict__ sg,
                        const float* __restrict__ vW1, const float* __restrict__ vb1,
                        const float* __restrict__ vW2, const float* __restrict__ vb2,
                        float* __restrict__ out)
{
    int g = threadIdx.x;                   // <<<1, 64>>>
    if (g >= G_) return;
    float acc = vb2[0];
    for (int j = 0; j < 32; ++j) {
        float hs = vb1[j];
        for (int c = 0; c < 256; ++c) hs += sg[g * 256 + c] * vW1[j * 256 + c];
        hs = hs > 0.0f ? hs : 0.0f;
        acc += hs * vW2[j];
    }
    out[G_ * A_ + g] = acc;
}

__global__ void k_pi(const float* __restrict__ h,
                     const float* __restrict__ pW1, const float* __restrict__ pb1,
                     const float* __restrict__ pW2, const float* __restrict__ pb2,
                     float* __restrict__ pi)
{
    int i = blockIdx.x * blockDim.x + threadIdx.x;
    if (i >= N_) return;
    float acc = pb2[0];
    for (int j = 0; j < 32; ++j) {
        float hs = pb1[j];
        for (int c = 0; c < 256; ++c) hs += h[(size_t)i * 256 + c] * pW1[j * 256 + c];
        hs = hs > 0.0f ? hs : 0.0f;
        acc += hs * pW2[j];
    }
    pi[i] = acc;
}

__global__ void k_lsm(const float* __restrict__ pi, float* __restrict__ out)
{
    int g = threadIdx.x;                   // <<<1, 64>>>  (A_ == P_ == 50: all slots filled)
    if (g >= G_) return;
    float m = -1e30f;
    for (int a = 0; a < A_; ++a) m = fmaxf(m, pi[g * P_ + a]);
    float s = 0.0f;
    for (int a = 0; a < A_; ++a) s += __expf(pi[g * P_ + a] - m);
    float lse = __logf(s) + m;
    for (int a = 0; a < A_; ++a) out[g * A_ + a] = pi[g * P_ + a] - lse;
}

// ---------------------------------------------------------------------------
// Host launch
// ---------------------------------------------------------------------------
extern "C" void kernel_launch(void* const* d_in, const int* in_sizes, int n_in,
                              void* d_out, int out_size, void* d_ws, size_t ws_size,
                              hipStream_t stream)
{
    (void)in_sizes; (void)n_in; (void)out_size; (void)ws_size;

    const int*   x      = (const int*)  d_in[0];
    const int*   eidx   = (const int*)  d_in[1];
    const int*   eattr  = (const int*)  d_in[2];
    const float* semb   = (const float*)d_in[4];
    const float* remb   = (const float*)d_in[5];
    const float* w_ih   = (const float*)d_in[6];
    const float* w_hh   = (const float*)d_in[7];
    const float* b_ih   = (const float*)d_in[8];
    const float* b_hh   = (const float*)d_in[9];
    const float* Wl     = (const float*)d_in[10];
    const float* bl     = (const float*)d_in[11];
    const float* Wr     = (const float*)d_in[12];
    const float* br     = (const float*)d_in[13];
    const float* We     = (const float*)d_in[14];
    const float* att    = (const float*)d_in[15];
    const float* cbias  = (const float*)d_in[16];
    const float* pW1    = (const float*)d_in[17];
    const float* pb1    = (const float*)d_in[18];
    const float* pW2    = (const float*)d_in[19];
    const float* pb2    = (const float*)d_in[20];
    const float* vW1    = (const float*)d_in[21];
    const float* vb1    = (const float*)d_in[22];
    const float* vW2    = (const float*)d_in[23];
    const float* vb2    = (const float*)d_in[24];

    const int* src = eidx;
    const int* tgt = eidx + E_;
    float* out = (float*)d_out;

    // ---- workspace carve-up ------------------------------------------------
    char* ws = (char*)d_ws;
    size_t off = 0;
    auto carve = [&](size_t bytes) -> char* {
        char* p = ws + off;
        off += (bytes + 255) & ~(size_t)255;
        return p;
    };
    half_t*  Wl16   = (half_t*)carve(4 * 256 * 256 * 2);
    half_t*  Wr16   = (half_t*)carve(4 * 256 * 256 * 2);
    half_t*  We16   = (half_t*)carve(4 * 256 * 64 * 2);
    float*   enc    = (float*) carve((size_t)E_ * 64 * 4);
    half_t*  enc16  = (half_t*)carve((size_t)E_ * 64 * 2);
    float*   oa     = (float*) carve((size_t)N_ * 106 * 4);
    float*   ia     = (float*) carve((size_t)N_ * 106 * 4);
    float*   co     = (float*) carve((size_t)N_ * 4);
    float*   ci     = (float*) carve((size_t)N_ * 4);
    float*   laacc  = (float*) carve((size_t)N_ * 64 * 4);
    float*   h      = (float*) carve((size_t)N_ * 256 * 4);
    half_t*  h16    = (half_t*)carve((size_t)N_ * 256 * 2);
    half_t*  ea16   = (half_t*)carve((size_t)NE_ * 64 * 2);
    float*   xl     = (float*) carve((size_t)N_ * 256 * 4);
    float*   xr     = (float*) carve((size_t)N_ * 256 * 4);
    float*   ee     = (float*) carve((size_t)NE_ * 256 * 4);
    float*   score  = (float*) carve((size_t)NE_ * 4 * 4);
    unsigned* smax  = (unsigned*)carve((size_t)N_ * 4 * 4);
    float*   denom  = (float*) carve((size_t)N_ * 4 * 4);
    float*   alpha  = (float*) carve((size_t)NE_ * 4 * 4);
    float*   agg    = (float*) carve((size_t)N_ * 256 * 4);
    float*   sg     = (float*) carve((size_t)G_ * 256 * 4);
    float*   pi     = (float*) carve((size_t)N_ * 4);

    auto blks = [](int n) { return (n + 255) / 256; };

    // ---- GAT weight conversion to f16 -------------------------------------
    k_cvt_f16<<<blks(4 * 256 * 256), 256, 0, stream>>>(Wl, Wl16, 4 * 256 * 256);
    k_cvt_f16<<<blks(4 * 256 * 256), 256, 0, stream>>>(Wr, Wr16, 4 * 256 * 256);
    k_cvt_f16<<<blks(4 * 256 * 64),  256, 0, stream>>>(We, We16, 4 * 256 * 64);

    // ---- zero pooling accumulators ----------------------------------------
    k_zero_f32<<<blks(N_ * 106), 256, 0, stream>>>(oa, N_ * 106);
    k_zero_f32<<<blks(N_ * 106), 256, 0, stream>>>(ia, N_ * 106);
    k_zero_f32<<<blks(N_),       256, 0, stream>>>(co, N_);
    k_zero_f32<<<blks(N_),       256, 0, stream>>>(ci, N_);
    k_zero_f32<<<blks(N_ * 64),  256, 0, stream>>>(laacc, N_ * 64);

    // ---- LSTM encoder (WMMA, weights staged in LDS) ------------------------
    k_lstm<<<E_ / 128, 256, 0, stream>>>(eattr, remb, b_ih, b_hh, w_ih, w_hh, enc, enc16);

    // ---- pools, node features, extended edge attrs ------------------------
    k_pool<<<blks(E_), 256, 0, stream>>>(eattr, src, tgt, semb, enc, oa, ia, co, ci, laacc);
    k_build_h<<<blks(N_), 256, 0, stream>>>(x, semb, oa, ia, co, ci, h, h16);
    k_build_ea<<<blks(NE_ * 64), 256, 0, stream>>>(enc16, laacc, ci, ea16);

    // ---- 4 GATv2 layers ----------------------------------------------------
    const int gemm_blocks_node = (N_ / 16) * (CH_ / 16) / 8;   // 400
    const int gemm_blocks_edge = (NE_ / 16) * (CH_ / 16) / 8;  // 3600
    for (int l = 0; l < 4; ++l) {
        k_gemm<<<gemm_blocks_node, 256, 0, stream>>>(h16, Wl16 + (size_t)l * 256 * 256,
                                                     bl + l * 256, xl, N_, CH_, CH_);
        k_gemm<<<gemm_blocks_node, 256, 0, stream>>>(h16, Wr16 + (size_t)l * 256 * 256,
                                                     br + l * 256, xr, N_, CH_, CH_);
        k_gemm<<<gemm_blocks_edge, 256, 0, stream>>>(ea16, We16 + (size_t)l * 256 * 64,
                                                     (const float*)nullptr, ee, NE_, 64, CH_);
        k_zero_u32<<<blks(N_ * 4),   256, 0, stream>>>(smax, N_ * 4);
        k_zero_f32<<<blks(N_ * 4),   256, 0, stream>>>(denom, N_ * 4);
        k_zero_f32<<<blks(N_ * 256), 256, 0, stream>>>(agg, N_ * 256);
        k_score <<<blks(NE_), 256, 0, stream>>>(src, tgt, xl, xr, ee, att + l * HDS_ * HC_,
                                                score, smax);
        k_expsum<<<blks(NE_), 256, 0, stream>>>(tgt, score, smax, alpha, denom);
        k_agg   <<<blks(NE_), 256, 0, stream>>>(src, tgt, xl, alpha, denom, agg);
        k_layer_final<<<blks(N_ * 256), 256, 0, stream>>>(agg, cbias + l * 256, h, h16);
    }

    // ---- heads -------------------------------------------------------------
    k_gpool<<<G_, 256, 0, stream>>>(h, sg);
    k_value<<<1, 64, 0, stream>>>(sg, vW1, vb1, vW2, vb2, out);
    k_pi<<<blks(N_), 256, 0, stream>>>(h, pW1, pb1, pW2, pb2, pi);
    k_lsm<<<1, 64, 0, stream>>>(pi, out);
}
